// CausalWanSelfAttention_27487790694418
// MI455X (gfx1250) — matrix operand order repaired and checked by reading, not compiled, not run on hardware
//
#include <hip/hip_runtime.h>
#include <hip/hip_bf16.h>
#include <math.h>

typedef __attribute__((ext_vector_type(16))) __bf16 v16bf;
typedef __attribute__((ext_vector_type(8)))  __bf16 v8bf;
typedef __attribute__((ext_vector_type(8)))  float  v8f;

#define S_LEN 1560
#define D_DIM 1536
#define NH    12
#define HDIM  128
#define PASTL 6240
#define KVLEN (PASTL + S_LEN)    /* 7800 */
#define W_G   52
#define EPSV  1e-6f
#define SM_SCALE 0.08838834764831845f /* 1/sqrt(128) */
#define QTILES ((S_LEN + 15) / 16)    /* 98 */

// ---------------------------------------------------------------------------
// GEMM: C[M x 1536] = A[M x K] @ W^T (+ bias), W is [1536 x K] row-major.
// One wave = 16 output cols x 64 output rows (4 M-tiles), K-step 32, bf16 WMMA.
// Edge rows handled by clamping the A address (branch-free; never stored).
// Prefetches are unconditional: speculative prefetch past the buffer end is
// silently dropped by hardware (ISA 10.5), so no guard branches are needed.
// ---------------------------------------------------------------------------
__global__ __launch_bounds__(256) void gemm_xwt_bf16(
    const float* __restrict__ A, const float* __restrict__ Wt,
    const float* __restrict__ bias,
    float* __restrict__ Cf, __bf16* __restrict__ Cb, int M, int K)
{
  const int lane = threadIdx.x & 31;
  const int wv   = threadIdx.x >> 5;
  const int half = lane >> 4;
  const int lr   = lane & 15;
  const int n0   = (blockIdx.x * 8 + wv) * 16;
  const int m0   = blockIdx.y * 64;
  const int n    = n0 + lr;

  v8f acc[4];
  {
    float bvl = bias[n];
    #pragma unroll
    for (int t = 0; t < 4; ++t)
      #pragma unroll
      for (int r = 0; r < 8; ++r) acc[t][r] = bvl;
  }

  // Clamped row pointers (garbage results in clamped rows are never stored).
  const float* aptr[4];
  #pragma unroll
  for (int t = 0; t < 4; ++t) {
    int row = m0 + t*16 + lr;
    if (row >= M) row = M - 1;
    aptr[t] = A + (size_t)row * K;
  }
  const float* wp0 = Wt + (size_t)n * K + half * 16;

  for (int k0 = 0; k0 < K; k0 += 32) {
    // B fragment: lane holds col n; VGPR j holds K = half*16 + 2j, 2j+1
    v16bf bfrag;
    {
      const float* wp = wp0 + k0;
      #pragma unroll
      for (int j = 0; j < 8; ++j) {
        bfrag[2*j]   = (__bf16)wp[2*j];
        bfrag[2*j+1] = (__bf16)wp[2*j+1];
      }
      __builtin_prefetch(wp + 32, 0, 0);
    }
    #pragma unroll
    for (int t = 0; t < 4; ++t) {
      const float* ap = aptr[t] + k0;
      v16bf afrag;
      #pragma unroll
      for (int j = 0; j < 8; ++j) {
        const int kk = half*16 + 2*j;
        afrag[2*j]   = (__bf16)ap[kk];
        afrag[2*j+1] = (__bf16)ap[kk+1];
      }
      __builtin_prefetch(ap + 32, 0, 0);
      acc[t] = __builtin_amdgcn_wmma_f32_16x16x32_bf16(
          false, afrag, false, bfrag, (short)0, acc[t], false, false);
    }
  }

  #pragma unroll
  for (int t = 0; t < 4; ++t)
    #pragma unroll
    for (int r = 0; r < 8; ++r) {
      const int row = m0 + t*16 + r + 8*half;
      if (row < M) {
        const size_t idx = (size_t)row * D_DIM + n;
        if (Cf) Cf[idx] = acc[t][r];
        if (Cb) Cb[idx] = (__bf16)acc[t][r];
      }
    }
}

// ---------------------------------------------------------------------------
// Elementwise fp32 -> bf16 conversion of past K/V cache.
// ---------------------------------------------------------------------------
__global__ void cvt_past(const float* __restrict__ pk, const float* __restrict__ pv,
                         __bf16* __restrict__ kb, __bf16* __restrict__ vb, int n)
{
  int i = blockIdx.x * blockDim.x + threadIdx.x;
  const int stride = gridDim.x * blockDim.x;
  for (; i < n; i += stride) {
    kb[i] = (__bf16)pk[i];
    vb[i] = (__bf16)pv[i];
  }
}

// ---------------------------------------------------------------------------
// RMSNorm + RoPE. grid=(S, 2): y=0 -> q path (bf16 out), y=1 -> k path
// (fp32 rk output + bf16 into the KV-cache buffer at offset PASTL).
// ---------------------------------------------------------------------------
__global__ __launch_bounds__(256) void rms_rope(
    const float* __restrict__ q_raw, const float* __restrict__ k_raw,
    const float* __restrict__ gq, const float* __restrict__ gk,
    const float* __restrict__ freqs, const int* __restrict__ start_frame,
    __bf16* __restrict__ rqb, float* __restrict__ rk_out, __bf16* __restrict__ kb)
{
  __shared__ float red[256];
  __shared__ float s_inv;
  const int s   = blockIdx.x;
  const int isK = blockIdx.y;
  const float* src = isK ? k_raw : q_raw;
  const float* g   = isK ? gk : gq;
  const float* row = src + (size_t)s * D_DIM;

  float ss = 0.f;
  for (int i = threadIdx.x; i < D_DIM; i += 256) { float v = row[i]; ss += v * v; }
  red[threadIdx.x] = ss;
  __syncthreads();
  for (int off = 128; off > 0; off >>= 1) {
    if ((int)threadIdx.x < off) red[threadIdx.x] += red[threadIdx.x + off];
    __syncthreads();
  }
  if (threadIdx.x == 0) s_inv = rsqrtf(red[0] / (float)D_DIM + EPSV);
  __syncthreads();
  const float inv = s_inv;

  const int sf = start_frame[0];
  const int h  = s / W_G;
  const int w  = s - h * W_G;
  for (int p = threadIdx.x; p < D_DIM / 2; p += 256) {
    const int head = p >> 6;
    const int i    = p & 63;
    float ang;
    if (i < 22)      ang = freqs[(size_t)sf * 64 + i];
    else if (i < 43) ang = freqs[(size_t)h  * 64 + i];
    else             ang = freqs[(size_t)w  * 64 + i];
    float sn, cs;
    __sincosf(ang, &sn, &cs);
    const int    d    = head * HDIM + 2 * i;
    const size_t base = (size_t)s * D_DIM + d;
    const float xr = row[d]     * inv * g[d];
    const float xi = row[d + 1] * inv * g[d + 1];
    const float outr = xr * cs - xi * sn;
    const float outi = xr * sn + xi * cs;
    if (isK) {
      rk_out[base]     = outr;
      rk_out[base + 1] = outi;
      const size_t kbase = (size_t)(PASTL + s) * D_DIM + d;
      kb[kbase]     = (__bf16)outr;
      kb[kbase + 1] = (__bf16)outi;
    } else {
      rqb[base]     = (__bf16)outr;
      rqb[base + 1] = (__bf16)outi;
    }
  }
}

// ---------------------------------------------------------------------------
// Flash attention. Block = 8 waves (one head, 8 query tiles of 16).
// Per 32-key block: K staged into LDS with ASYNC global->LDS b128 copies
// (no VGPR roundtrip, waited with s_wait_asynccnt), V staged transposed with
// b128 loads + b16 scatters. Wave computes S^T = K_blk @ Q^T (lane = query
// column -> softmax stats per-lane + one shfl_xor(16)), rebuilds P^T via
// cross-half shuffles, accumulates O^T = V^T @ P^T.
// ---------------------------------------------------------------------------
__global__ __launch_bounds__(256) void flash_attn(
    const __bf16* __restrict__ rqb, const __bf16* __restrict__ kb,
    const __bf16* __restrict__ vb, float* __restrict__ o_ws)
{
  __shared__ __bf16 kS[32][136];    // [key][hd], row pad 16B -> conflict-free
  __shared__ __bf16 vT[HDIM][34];   // [hd][key], padded
  const int lane = threadIdx.x & 31;
  const int wv   = threadIdx.x >> 5;
  const int half = lane >> 4;
  const int lr   = lane & 15;
  const int head = blockIdx.y;

  int qt = blockIdx.x * 8 + wv;
  const bool qvalid = (qt < QTILES);
  if (!qvalid) qt = QTILES - 1;            // duplicate work, no store
  const int q0   = qt * 16;
  const int qrow = q0 + lr;

  const unsigned kS_lds = (unsigned)(uintptr_t)&kS[0][0];

  // Q B-fragments (lane = query col; K along head-dim, 4 steps of 32)
  v16bf qf[4];
  {
    const int qclamp = (qrow < S_LEN) ? qrow : (S_LEN - 1);
    const __bf16* qp = rqb + (size_t)qclamp * D_DIM + head * HDIM;
    const bool rv = (qrow < S_LEN);
    #pragma unroll
    for (int d = 0; d < 4; ++d)
      #pragma unroll
      for (int j = 0; j < 8; ++j) {
        const int kk = d * 32 + half * 16 + 2 * j;
        qf[d][2*j]   = rv ? qp[kk]   : (__bf16)0.0f;
        qf[d][2*j+1] = rv ? qp[kk+1] : (__bf16)0.0f;
      }
  }

  v8f ot[8];
  #pragma unroll
  for (int t = 0; t < 8; ++t)
    #pragma unroll
    for (int r = 0; r < 8; ++r) ot[t][r] = 0.f;
  float row_max = -__builtin_inff();
  float row_sum = 0.f;

  const int nkb = (KVLEN + 31) / 32;
  for (int kbI = 0; kbI < nkb; ++kbI) {
    const int k_base = kbI * 32;

    // --- K staging: async global->LDS, 512 x 16B chunks over 256 threads ---
    #pragma unroll
    for (int i = 0; i < 2; ++i) {
      const int c   = threadIdx.x + 256 * i;
      const int key = c >> 4;
      const int hcb = (c & 15) * 8;
      int krow = k_base + key;
      if (krow >= KVLEN) krow = KVLEN - 1;   // garbage keys masked later
      const __bf16* gp = kb + (size_t)krow * D_DIM + head * HDIM + hcb;
      const unsigned loff = kS_lds + (unsigned)(key * 136 + hcb) * 2u;
      asm volatile("global_load_async_to_lds_b128 %0, %1, off"
                   :: "v"(loff), "v"(gp) : "memory");
    }
    // --- V staging: b128 loads + transposed b16 scatter into LDS ---
    #pragma unroll
    for (int i = 0; i < 2; ++i) {
      const int c   = threadIdx.x + 256 * i;
      const int key = c >> 4;
      const int hcb = (c & 15) * 8;
      int krow = k_base + key;
      if (krow >= KVLEN) krow = KVLEN - 1;
      const v8bf vv = *(const v8bf*)(vb + (size_t)krow * D_DIM + head * HDIM + hcb);
      #pragma unroll
      for (int e = 0; e < 8; ++e) vT[hcb + e][key] = vv[e];
    }
    asm volatile("s_wait_asynccnt 0x0" ::: "memory");
    __syncthreads();

    // --- scores^T: two 16(keys) x 16(queries) tiles, K frags from LDS ---
    v8f sc0, sc1;
    #pragma unroll
    for (int r = 0; r < 8; ++r) { sc0[r] = 0.f; sc1[r] = 0.f; }
    {
      const __bf16* kr0 = &kS[lr][0];
      const __bf16* kr1 = &kS[16 + lr][0];
      #pragma unroll
      for (int d = 0; d < 4; ++d) {
        v16bf ka0, ka1;
        #pragma unroll
        for (int j = 0; j < 8; ++j) {
          const int kk = d * 32 + half*16 + 2*j;
          ka0[2*j]   = kr0[kk];  ka0[2*j+1] = kr0[kk+1];
          ka1[2*j]   = kr1[kk];  ka1[2*j+1] = kr1[kk+1];
        }
        sc0 = __builtin_amdgcn_wmma_f32_16x16x32_bf16(false, ka0, false, qf[d], (short)0, sc0, false, false);
        sc1 = __builtin_amdgcn_wmma_f32_16x16x32_bf16(false, ka1, false, qf[d], (short)0, sc1, false, false);
      }
    }

    // --- online softmax (per lane = per query; combine halves via shfl) ---
    float p0a[8], p1a[8];
    float mloc = -__builtin_inff();
    #pragma unroll
    for (int r = 0; r < 8; ++r) {
      const int key0 = k_base + 8*half + r;
      const int key1 = key0 + 16;
      const float s0v = (key0 < KVLEN) ? sc0[r] * SM_SCALE : -__builtin_inff();
      const float s1v = (key1 < KVLEN) ? sc1[r] * SM_SCALE : -__builtin_inff();
      p0a[r] = s0v; p1a[r] = s1v;
      mloc = fmaxf(mloc, fmaxf(s0v, s1v));
    }
    mloc = fmaxf(mloc, __shfl_xor(mloc, 16));
    const float m_new = fmaxf(row_max, mloc);
    const float alpha = __expf(row_max - m_new);
    float bs = 0.f;
    #pragma unroll
    for (int r = 0; r < 8; ++r) {
      p0a[r] = __expf(p0a[r] - m_new);
      p1a[r] = __expf(p1a[r] - m_new);
      bs += p0a[r] + p1a[r];
    }
    bs += __shfl_xor(bs, 16);
    row_sum = row_sum * alpha + bs;
    row_max = m_new;
    #pragma unroll
    for (int t = 0; t < 8; ++t)
      #pragma unroll
      for (int r = 0; r < 8; ++r) ot[t][r] *= alpha;

    // --- rebuild P^T as a B-fragment (32 keys x 16 queries) ---
    float q0s[8], q1s[8];
    #pragma unroll
    for (int r = 0; r < 8; ++r) {
      q0s[r] = __shfl_xor(p0a[r], 16);
      q1s[r] = __shfl_xor(p1a[r], 16);
    }
    v16bf pf;
    #pragma unroll
    for (int j = 0; j < 4; ++j) {             // K rows 2j,2j+1 of tile[half]
      const float e0 = half ? q1s[2*j]   : p0a[2*j];
      const float e1 = half ? q1s[2*j+1] : p0a[2*j+1];
      pf[2*j] = (__bf16)e0; pf[2*j+1] = (__bf16)e1;
    }
    #pragma unroll
    for (int j = 4; j < 8; ++j) {             // K rows 2j,2j+1 (upper half)
      const float e0 = half ? p1a[2*j-8] : q0s[2*j-8];
      const float e1 = half ? p1a[2*j-7] : q0s[2*j-7];
      pf[2*j] = (__bf16)e0; pf[2*j+1] = (__bf16)e1;
    }

    // --- O^T += V^T @ P^T  (8 hd-tiles) ---
    #pragma unroll
    for (int t = 0; t < 8; ++t) {
      v16bf vf;
      const int hd = t * 16 + lr;
      #pragma unroll
      for (int j = 0; j < 8; ++j) {
        const int kk = half*16 + 2*j;
        vf[2*j]   = vT[hd][kk];
        vf[2*j+1] = vT[hd][kk+1];
      }
      ot[t] = __builtin_amdgcn_wmma_f32_16x16x32_bf16(false, vf, false, pf, (short)0, ot[t], false, false);
    }
    __syncthreads();
  }

  // epilogue: normalize and store O (lane = query col, rows = hd)
  const float invs = (row_sum > 0.f) ? (1.f / row_sum) : 0.f;
  if (qvalid && qrow < S_LEN) {
    float* op = o_ws + (size_t)qrow * D_DIM + head * HDIM;
    #pragma unroll
    for (int t = 0; t < 8; ++t)
      #pragma unroll
      for (int r = 0; r < 8; ++r)
        op[t*16 + r + 8*half] = ot[t][r] * invs;
  }
}

// ---------------------------------------------------------------------------
extern "C" void kernel_launch(void* const* d_in, const int* in_sizes, int n_in,
                              void* d_out, int out_size, void* d_ws, size_t ws_size,
                              hipStream_t stream)
{
  (void)in_sizes; (void)n_in; (void)out_size; (void)ws_size;

  const float* x      = (const float*)d_in[0];
  const float* freqs  = (const float*)d_in[3];
  const float* past_k = (const float*)d_in[4];
  const float* past_v = (const float*)d_in[5];
  const float* Wq = (const float*)d_in[6];
  const float* bq = (const float*)d_in[7];
  const float* Wk = (const float*)d_in[8];
  const float* bk = (const float*)d_in[9];
  const float* Wv = (const float*)d_in[10];
  const float* bv = (const float*)d_in[11];
  const float* Wo = (const float*)d_in[12];
  const float* bo = (const float*)d_in[13];
  const float* gq = (const float*)d_in[14];
  const float* gk = (const float*)d_in[15];
  const int* start_frame = (const int*)d_in[16];

  float* y_out  = (float*)d_out;
  float* rk_out = y_out  + (size_t)S_LEN * D_DIM;
  float* v_out  = rk_out + (size_t)S_LEN * D_DIM;

  char* ws = (char*)d_ws;
  size_t off = 0;
  auto alloc = [&](size_t bytes) -> char* {
    char* p = ws + off;
    off += (bytes + 255) & ~(size_t)255;
    return p;
  };
  float*  q_raw = (float*) alloc((size_t)S_LEN * D_DIM * 4);
  float*  k_raw = (float*) alloc((size_t)S_LEN * D_DIM * 4);
  float*  o_ws  = (float*) alloc((size_t)S_LEN * D_DIM * 4);
  __bf16* rqb   = (__bf16*)alloc((size_t)S_LEN * D_DIM * 2);
  __bf16* kbf   = (__bf16*)alloc((size_t)KVLEN * D_DIM * 2);
  __bf16* vbf   = (__bf16*)alloc((size_t)KVLEN * D_DIM * 2);

  const dim3 gG(12, (S_LEN + 63) / 64, 1);   // 12 x 25
  const dim3 tG(256);

  // QKV projections (V writes fp32 output + bf16 into KV cache tail)
  gemm_xwt_bf16<<<gG, tG, 0, stream>>>(x, Wq, bq, q_raw, (__bf16*)nullptr, S_LEN, D_DIM);
  gemm_xwt_bf16<<<gG, tG, 0, stream>>>(x, Wk, bk, k_raw, (__bf16*)nullptr, S_LEN, D_DIM);
  gemm_xwt_bf16<<<gG, tG, 0, stream>>>(x, Wv, bv, v_out, vbf + (size_t)PASTL * D_DIM, S_LEN, D_DIM);

  // KV cache fp32 -> bf16
  cvt_past<<<2048, 256, 0, stream>>>(past_k, past_v, kbf, vbf, PASTL * D_DIM);

  // RMSNorm + RoPE (q -> bf16 ws, k -> fp32 out + bf16 cache tail)
  rms_rope<<<dim3(S_LEN, 2), 256, 0, stream>>>(q_raw, k_raw, gq, gk, freqs,
                                               start_frame, rqb, rk_out, kbf);

  // Flash attention: grid = (ceil(98/8), heads), 8 waves/block
  flash_attn<<<dim3((QTILES + 7) / 8, NH), 256, 0, stream>>>(rqb, kbf, vbf, o_ws);

  // Output projection
  gemm_xwt_bf16<<<gG, tG, 0, stream>>>(o_ws, Wo, bo, y_out, (__bf16*)nullptr, S_LEN, D_DIM);
}
